// CrossModalTransformerBlock_41729902248483
// MI455X (gfx1250) — matrix operand order repaired and checked by reading, not compiled, not run on hardware
//
#include <hip/hip_runtime.h>
#include <math.h>

// ---------------------------------------------------------------------------
// CDNA5 (gfx1250) cross-modal transformer block.
// Matmuls via v_wmma_f32_16x16x32_bf16 (wave32); global->LDS staging via
// GLOBAL_LOAD_ASYNC_TO_LDS_B128 + ASYNCcnt double buffering when available.
// fp32 residual streams, bf16 matrix operands, f32 accumulation.
// ---------------------------------------------------------------------------

typedef unsigned short u16t;
typedef unsigned int   u32t;
typedef __bf16 v16bf __attribute__((ext_vector_type(16)));
typedef float  v8f   __attribute__((ext_vector_type(8)));
typedef int    v4i   __attribute__((vector_size(16)));   // int4, builtin's pointee

union FragBF { v16bf v; uint4 q[2]; };
union U4x8  { uint4 q; u16t s[8]; };

#define MODE_BF      0   // out_bf16 = acc + bias
#define MODE_GELU    1   // out_bf16 = gelu(acc + bias)
#define MODE_RES     2   // out_f32  = res + acc + bias
#define MODE_HALFRES 3   // out_f32  = res + 0.5*(acc + bias)

#if __has_builtin(__builtin_amdgcn_global_load_async_to_lds_b128)
#define HAS_ASYNC_LDS 1
// builtin signature (from clang diagnostics): (v4i AS1*, v4i AS3*, i32 off, i32 cpol)
#define GASYNC16(dst_lds, src_g)                                            \
  __builtin_amdgcn_global_load_async_to_lds_b128(                           \
      (v4i __attribute__((address_space(1)))*)(src_g),                      \
      (v4i __attribute__((address_space(3)))*)(dst_lds), 0, 0)
#else
#define HAS_ASYNC_LDS 0
#endif

#if __has_builtin(__builtin_amdgcn_s_wait_asynccnt)
#define WAIT_ASYNC(n) __builtin_amdgcn_s_wait_asynccnt(n)
#else
#define WAIT_ASYNC(n) asm volatile("s_wait_asynccnt %0" ::"i"(n) : "memory")
#endif

__device__ __forceinline__ u16t f2bf(float f) {
  u32t u = __float_as_uint(f);
  u32t r = u + 0x7FFFu + ((u >> 16) & 1u);   // round-to-nearest-even
  return (u16t)(r >> 16);
}

__device__ __forceinline__ v8f wmma_bf16(v16bf a, v16bf b, v8f c) {
  // 8 args: (neg_a, A, neg_b, B, c_mod, C, reuse_a, reuse_b)
  return __builtin_amdgcn_wmma_f32_16x16x32_bf16(false, a, false, b, (short)0, c,
                                                 false, false);
}

// ---------------------------------------------------------------------------
// Weight convert+transpose: W[K][N] f32 -> Wt[N][K] bf16
// ---------------------------------------------------------------------------
__global__ __launch_bounds__(256) void wconv_kernel(const float* __restrict__ W,
                                                    u16t* __restrict__ Wt,
                                                    int K, int N) {
  int idx = blockIdx.x * 256 + threadIdx.x;
  if (idx < K * N) {
    int n = idx / K;
    int k = idx - n * K;
    Wt[idx] = f2bf(W[(size_t)k * N + n]);
  }
}

// ---------------------------------------------------------------------------
// LayerNorm (D=768): fp32 in -> bf16 out. One row per block.
// ---------------------------------------------------------------------------
__global__ __launch_bounds__(256) void ln_kernel(const float* __restrict__ X,
                                                 const float* __restrict__ g,
                                                 const float* __restrict__ b,
                                                 u16t* __restrict__ out) {
  const int row = blockIdx.x;
  const int tid = threadIdx.x;
  const float* xr = X + (size_t)row * 768;
  float s = 0.f, s2 = 0.f;
  for (int i = tid; i < 768; i += 256) { float v = xr[i]; s += v; s2 += v * v; }
  __shared__ float r1[256];
  __shared__ float r2[256];
  r1[tid] = s; r2[tid] = s2;
  __syncthreads();
  for (int st = 128; st > 0; st >>= 1) {
    if (tid < st) { r1[tid] += r1[tid + st]; r2[tid] += r2[tid + st]; }
    __syncthreads();
  }
  float mu  = r1[0] * (1.0f / 768.0f);
  float var = r2[0] * (1.0f / 768.0f) - mu * mu;
  float inv = rsqrtf(var + 1e-5f);
  for (int i = tid; i < 768; i += 256)
    out[(size_t)row * 768 + i] = f2bf((xr[i] - mu) * inv * g[i] + b[i]);
}

// ---------------------------------------------------------------------------
// GEMM: out[M,N] = A[M,K](bf16) * Wt[N,K]^T(bf16) + bias, fused epilogue.
// Block = 256 thr = 8 waves; block tile 128x128; wave tile 32x64
// (2x4 grid of 16x16 WMMA accumulators). K stepped in chunks of 32.
// Staging: async global->LDS (ASYNCcnt) with LDS double buffering.
// M,N multiples of 128; K multiple of 32 (true for 768/3072 here).
// ---------------------------------------------------------------------------
__global__ __launch_bounds__(256) void gemm_bf16_kernel(
    const u16t* __restrict__ A,    // [M,K]
    const u16t* __restrict__ Bt,   // [N,K]
    const float* __restrict__ bias,
    const float* __restrict__ res, // residual (modes 2,3)
    float* __restrict__ outF,
    u16t* __restrict__ outB,
    int N, int K, int mode) {
#if HAS_ASYNC_LDS
  __shared__ __align__(16) u16t lA[2][128 * 32];
  __shared__ __align__(16) u16t lB[2][128 * 32];
#else
  __shared__ __align__(16) u16t lA[1][128 * 32];
  __shared__ __align__(16) u16t lB[1][128 * 32];
#endif

  const int tid  = threadIdx.x;
  const int lane = tid & 31;
  const int wave = tid >> 5;
  const int wm = wave >> 1;         // 0..3 : 32-row sub-tile
  const int wn = wave & 1;          // 0..1 : 64-col sub-tile
  const int hf = lane >> 4;         // wave half
  const int nl = lane & 15;
  const int m0 = blockIdx.y * 128;
  const int n0 = blockIdx.x * 128;
  const int cr = tid >> 1;          // copy row 0..127
  const int ck = (tid & 1) * 16;    // copy k offset 0/16

  const u16t* gA = A  + (size_t)(m0 + cr) * K + ck;
  const u16t* gB = Bt + (size_t)(n0 + cr) * K + ck;
  const int dst = cr * 32 + ck;

  const v8f vz = {0.f, 0.f, 0.f, 0.f, 0.f, 0.f, 0.f, 0.f};
  v8f acc[2][4];
#pragma unroll
  for (int i = 0; i < 2; i++)
#pragma unroll
    for (int j = 0; j < 4; j++) acc[i][j] = vz;

  const int nt = K >> 5;

#if HAS_ASYNC_LDS
  // ---- async double-buffered staging: 4 x b128 per thread per tile ----
  {
    GASYNC16(&lA[0][dst],     gA);
    GASYNC16(&lA[0][dst + 8], gA + 8);
    GASYNC16(&lB[0][dst],     gB);
    GASYNC16(&lB[0][dst + 8], gB + 8);
  }
  for (int t = 0; t < nt; t++) {
    const int buf = t & 1;
    if (t + 1 < nt) {
      const int k1 = (t + 1) << 5;
      GASYNC16(&lA[buf ^ 1][dst],     gA + k1);
      GASYNC16(&lA[buf ^ 1][dst + 8], gA + k1 + 8);
      GASYNC16(&lB[buf ^ 1][dst],     gB + k1);
      GASYNC16(&lB[buf ^ 1][dst + 8], gB + k1 + 8);
      WAIT_ASYNC(4);   // tile t complete (in-order), tile t+1 may remain
    } else {
      WAIT_ASYNC(0);
    }
    __syncthreads();

    FragBF fa0, fa1;
    {
      const u16t* p = &lA[buf][(wm * 32 + nl) * 32];
      fa0.q[0] = *(const uint4*)(p + hf * 16);
      fa0.q[1] = *(const uint4*)(p + hf * 16 + 8);
    }
    {
      const u16t* p = &lA[buf][(wm * 32 + 16 + nl) * 32];
      fa1.q[0] = *(const uint4*)(p + hf * 16);
      fa1.q[1] = *(const uint4*)(p + hf * 16 + 8);
    }
#pragma unroll
    for (int j = 0; j < 4; j++) {
      FragBF fb;
      const u16t* p = &lB[buf][(wn * 64 + j * 16 + nl) * 32 + hf * 16];
      fb.q[0] = *(const uint4*)p;
      fb.q[1] = *(const uint4*)(p + 8);
      acc[0][j] = wmma_bf16(fa0.v, fb.v, acc[0][j]);
      acc[1][j] = wmma_bf16(fa1.v, fb.v, acc[1][j]);
    }
    __syncthreads();   // buffer reuse fence (tile t+2 overwrites buf)
  }
#else
  // ---- synchronous fallback staging ----
  for (int t = 0; t < nt; t++) {
    const int k0 = t << 5;
    __syncthreads();
    {
      const uint4* pa = (const uint4*)(gA + k0);
      const uint4* pb = (const uint4*)(gB + k0);
      uint4 a0 = pa[0], a1 = pa[1];
      uint4 b0 = pb[0], b1 = pb[1];
      *(uint4*)&lA[0][dst]     = a0;
      *(uint4*)&lA[0][dst + 8] = a1;
      *(uint4*)&lB[0][dst]     = b0;
      *(uint4*)&lB[0][dst + 8] = b1;
    }
    if (k0 + 32 < K) {
      __builtin_prefetch(gA + k0 + 32, 0, 1);
      __builtin_prefetch(gB + k0 + 32, 0, 1);
    }
    __syncthreads();

    FragBF fa0, fa1;
    {
      const u16t* p = &lA[0][(wm * 32 + nl) * 32];
      fa0.q[0] = *(const uint4*)(p + hf * 16);
      fa0.q[1] = *(const uint4*)(p + hf * 16 + 8);
    }
    {
      const u16t* p = &lA[0][(wm * 32 + 16 + nl) * 32];
      fa1.q[0] = *(const uint4*)(p + hf * 16);
      fa1.q[1] = *(const uint4*)(p + hf * 16 + 8);
    }
#pragma unroll
    for (int j = 0; j < 4; j++) {
      FragBF fb;
      const u16t* p = &lB[0][(wn * 64 + j * 16 + nl) * 32 + hf * 16];
      fb.q[0] = *(const uint4*)p;
      fb.q[1] = *(const uint4*)(p + 8);
      acc[0][j] = wmma_bf16(fa0.v, fb.v, acc[0][j]);
      acc[1][j] = wmma_bf16(fa1.v, fb.v, acc[1][j]);
    }
  }
#endif

  // Epilogue. C layout: VGPR v -> row v + 8*hf, lane nl -> col.
#pragma unroll
  for (int i = 0; i < 2; i++)
#pragma unroll
    for (int j = 0; j < 4; j++) {
      int col = n0 + wn * 64 + j * 16 + nl;
      float bv = bias[col];
#pragma unroll
      for (int v = 0; v < 8; v++) {
        int row = m0 + wm * 32 + i * 16 + v + 8 * hf;
        size_t off = (size_t)row * N + col;
        float x = acc[i][j][v] + bv;
        if (mode == MODE_BF) {
          outB[off] = f2bf(x);
        } else if (mode == MODE_GELU) {
          float gx = 0.5f * x * (1.0f + erff(x * 0.70710678f));
          outB[off] = f2bf(gx);
        } else if (mode == MODE_RES) {
          outF[off] = res[off] + x;
        } else {
          outF[off] = res[off] + 0.5f * x;
        }
      }
    }
}

// ---------------------------------------------------------------------------
// Flash attention, hd=64, H=12. Grid (Tq/64, H, B); block 128 = 4 waves,
// each wave owns 16 query rows. S stepped in chunks of 32 keys:
// scores = 4 wmma, PV = 4 wmma per chunk. Online softmax with shuffle
// reductions across the 16-lane wave halves (C-layout rows live per-half).
// Q/K/V/O are bf16 [B*T, 768] with head h in columns h*64..h*64+63.
// ---------------------------------------------------------------------------
__global__ __launch_bounds__(128) void attn_kernel(const u16t* __restrict__ Q,
                                                   const u16t* __restrict__ Km,
                                                   const u16t* __restrict__ Vm,
                                                   u16t* __restrict__ O,
                                                   int T, int S) {
  __shared__ __align__(16) u16t VT[64 * 32];       // V chunk transposed [col][s]
  __shared__ __align__(16) u16t PS[4 * 16 * 32];   // per-wave P tiles

  const int b = blockIdx.z, h = blockIdx.y, mt = blockIdx.x;
  const int tid = threadIdx.x, lane = tid & 31, wave = tid >> 5;
  const int hf = lane >> 4, nl = lane & 15;
  const int hc = h * 64;
  const int qrow0 = b * T + mt * 64 + wave * 16;
  const int krow0 = b * S;

  // Q fragments: 16x64 = two 16x32 A-fragments, loaded straight from global.
  FragBF qf[2];
  {
    const u16t* qp = Q + (size_t)(qrow0 + nl) * 768 + hc;
    qf[0].q[0] = *(const uint4*)(qp + hf * 16);
    qf[0].q[1] = *(const uint4*)(qp + hf * 16 + 8);
    qf[1].q[0] = *(const uint4*)(qp + 32 + hf * 16);
    qf[1].q[1] = *(const uint4*)(qp + 32 + hf * 16 + 8);
  }

  const v8f vz = {0.f, 0.f, 0.f, 0.f, 0.f, 0.f, 0.f, 0.f};
  float rm[8], rs[8];
  v8f oacc[4];
#pragma unroll
  for (int v = 0; v < 8; v++) { rm[v] = -1e30f; rs[v] = 0.f; }
#pragma unroll
  for (int t = 0; t < 4; t++) oacc[t] = vz;

  u16t* Pw = &PS[wave * 16 * 32];

  for (int s0 = 0; s0 < S; s0 += 32) {
    // Stage V chunk transposed: VT[c][s] = V[s0+s][hc+c].
    // 16B global loads (8 head-cols of one key row), scattered ds_store_b16.
    __syncthreads();
    for (int i = tid; i < 256; i += 128) {
      int sl = i & 31;
      int c8 = (i >> 5) << 3;    // 0,8,...,56
      U4x8 u;
      u.q = *(const uint4*)(Vm + (size_t)(krow0 + s0 + sl) * 768 + hc + c8);
#pragma unroll
      for (int e = 0; e < 8; e++) VT[(c8 + e) * 32 + sl] = u.s[e];
    }
    __syncthreads();

    // scores (two 16x16 tiles covering 32 key columns), K=64 in 2 wmma each
    v8f sc0 = vz, sc1 = vz;
#pragma unroll
    for (int c = 0; c < 2; c++) {
      FragBF kf;
      const u16t* kp0 =
          Km + (size_t)(krow0 + s0 + nl) * 768 + hc + c * 32 + hf * 16;
      kf.q[0] = *(const uint4*)kp0;
      kf.q[1] = *(const uint4*)(kp0 + 8);
      sc0 = wmma_bf16(qf[c].v, kf.v, sc0);
      const u16t* kp1 =
          Km + (size_t)(krow0 + s0 + 16 + nl) * 768 + hc + c * 32 + hf * 16;
      kf.q[0] = *(const uint4*)kp1;
      kf.q[1] = *(const uint4*)(kp1 + 8);
      sc1 = wmma_bf16(qf[c].v, kf.v, sc1);
    }

    // online softmax; row for element v is (v + 8*hf), spread over 16 lanes
#pragma unroll
    for (int v = 0; v < 8; v++) {
      float x0 = sc0[v] * 0.125f;     // 1/sqrt(64)
      float x1 = sc1[v] * 0.125f;
      float t = fmaxf(x0, x1);
      for (int off = 1; off < 16; off <<= 1) t = fmaxf(t, __shfl_xor(t, off, 32));
      float nm = fmaxf(rm[v], t);
      float corr = __expf(rm[v] - nm);
      float p0 = __expf(x0 - nm), p1 = __expf(x1 - nm);
      float ps = p0 + p1;
      for (int off = 1; off < 16; off <<= 1) ps += __shfl_xor(ps, off, 32);
      rs[v] = rs[v] * corr + ps;
      rm[v] = nm;
#pragma unroll
      for (int t4 = 0; t4 < 4; t4++) oacc[t4][v] *= corr;
      int r = v + 8 * hf;
      Pw[r * 32 + nl]      = f2bf(p0);
      Pw[r * 32 + 16 + nl] = f2bf(p1);
    }
    __syncthreads();   // P: C-layout -> A-layout crosses lanes via LDS

    // P (16x32) as A fragment
    FragBF pf;
    const u16t* pp = &Pw[nl * 32];
    pf.q[0] = *(const uint4*)(pp + hf * 16);
    pf.q[1] = *(const uint4*)(pp + hf * 16 + 8);

    // out += P * V   (4 column tiles of 16)
#pragma unroll
    for (int t4 = 0; t4 < 4; t4++) {
      FragBF vf;
      const u16t* vp = &VT[(t4 * 16 + nl) * 32 + hf * 16];
      vf.q[0] = *(const uint4*)vp;
      vf.q[1] = *(const uint4*)(vp + 8);
      oacc[t4] = wmma_bf16(pf.v, vf.v, oacc[t4]);
    }
  }

  // normalize and store bf16
#pragma unroll
  for (int t4 = 0; t4 < 4; t4++)
#pragma unroll
    for (int v = 0; v < 8; v++) {
      int rg  = qrow0 + v + 8 * hf;
      int col = hc + t4 * 16 + nl;
      O[(size_t)rg * 768 + col] = f2bf(oacc[t4][v] / rs[v]);
    }
}

// ---------------------------------------------------------------------------
// Host orchestration
// ---------------------------------------------------------------------------
namespace {
struct Lin  { const float* w; const float* b; };
struct LNP  { const float* g; const float* b; };
struct MHA  { Lin q, k, v, o; };
struct FFNP { Lin l1, l2; };
struct ENC  { MHA a; FFNP f; LNP ln1, ln2; };
struct CRS  { MHA a; LNP ln1, ln2; };
}  // namespace

extern "C" void kernel_launch(void* const* d_in, const int* in_sizes, int n_in,
                              void* d_out, int out_size, void* d_ws,
                              size_t ws_size, hipStream_t stream) {
  (void)in_sizes; (void)n_in; (void)out_size; (void)ws_size;

  const float* visual = (const float*)d_in[0];  // [8,512,768]
  const float* audio  = (const float*)d_in[1];  // [8,512,768]
  const float* hr     = (const float*)d_in[2];  // [8,256,768]

  int idx = 3;
  auto P     = [&]() { return (const float*)d_in[idx++]; };
  auto rdLin = [&]() { Lin l; l.w = P(); l.b = P(); return l; };
  auto rdLN  = [&]() { LNP l; l.g = P(); l.b = P(); return l; };
  auto rdMHA = [&]() { MHA m; m.q = rdLin(); m.k = rdLin(); m.v = rdLin(); m.o = rdLin(); return m; };
  auto rdFFN = [&]() { FFNP f; f.l1 = rdLin(); f.l2 = rdLin(); return f; };
  auto rdENC = [&]() { ENC e; e.a = rdMHA(); e.f = rdFFN(); e.ln1 = rdLN(); e.ln2 = rdLN(); return e; };
  auto rdCRS = [&]() { CRS c; c.a = rdMHA(); c.ln1 = rdLN(); c.ln2 = rdLN(); return c; };

  ENC sav = rdENC(), saa = rdENC(), sah = rdENC();
  CRS v2a = rdCRS(), v2h = rdCRS(), a2v = rdCRS(), a2h = rdCRS(),
      h2v = rdCRS(), h2a = rdCRS();
  FFNP fv = rdFFN(), fa = rdFFN(), fh = rdFFN();
  LNP nv = rdLN(), na = rdLN(), nh = rdLN();

  // ----- workspace carve (~93 MB) -----
  char* wp = (char*)d_ws;
  auto carve = [&](size_t bytes) {
    void* p = wp;
    wp += (bytes + 255) & ~(size_t)255;
    return p;
  };
  u16t*  Wt  = (u16t*)carve(768ull * 3072 * 2);   // bf16 transposed weights
  u16t*  XN  = (u16t*)carve(4096ull * 768 * 2);   // LN out / attn out
  u16t*  XN2 = (u16t*)carve(4096ull * 768 * 2);   // LN out (kv)
  u16t*  Qb  = (u16t*)carve(4096ull * 768 * 2);
  u16t*  Kb  = (u16t*)carve(4096ull * 768 * 2);
  u16t*  Vb  = (u16t*)carve(4096ull * 768 * 2);
  u16t*  H1  = (u16t*)carve(4096ull * 3072 * 2);  // FFN hidden (bf16)
  float* Sv  = (float*)carve(4096ull * 768 * 4);  // residual streams
  float* Sa  = (float*)carve(4096ull * 768 * 4);
  float* Sh  = (float*)carve(2048ull * 768 * 4);

  const size_t vN = 8ull * 512 * 768;   // 3,145,728
  const size_t hN = 8ull * 256 * 768;

  auto gemm = [&](const u16t* Ab, Lin lin, int K, int N, int M, int mode,
                  const float* res, float* outF, u16t* outB) {
    int total = K * N;
    wconv_kernel<<<(total + 255) / 256, 256, 0, stream>>>(lin.w, Wt, K, N);
    dim3 grid(N / 128, M / 128);
    gemm_bf16_kernel<<<grid, 256, 0, stream>>>(Ab, Wt, lin.b, res, outF, outB,
                                               N, K, mode);
  };
  auto lnorm = [&](const float* X, LNP p, u16t* out, int M) {
    ln_kernel<<<M, 256, 0, stream>>>(X, p.g, p.b, out);
  };
  auto attn = [&](const u16t* Qp, const u16t* Kp, const u16t* Vp, u16t* Op,
                  int Tq, int Tkv) {
    dim3 grid(Tq / 64, 12, 8);
    attn_kernel<<<grid, 128, 0, stream>>>(Qp, Kp, Vp, Op, Tq, Tkv);
  };

  auto encoder = [&](float* S, int T, const ENC& e) {
    int M = 8 * T;
    lnorm(S, e.ln1, XN, M);
    gemm(XN, e.a.q, 768, 768, M, MODE_BF, nullptr, nullptr, Qb);
    gemm(XN, e.a.k, 768, 768, M, MODE_BF, nullptr, nullptr, Kb);
    gemm(XN, e.a.v, 768, 768, M, MODE_BF, nullptr, nullptr, Vb);
    attn(Qb, Kb, Vb, XN, T, T);                    // XN reused as attn-out
    gemm(XN, e.a.o, 768, 768, M, MODE_RES, S, S, nullptr);
    lnorm(S, e.ln2, XN, M);
    gemm(XN, e.f.l1, 768, 3072, M, MODE_GELU, nullptr, nullptr, H1);
    gemm(H1, e.f.l2, 3072, 768, M, MODE_RES, S, S, nullptr);
  };

  auto cross = [&](const float* Sq, int Tq, const float* Skv, int Tkv,
                   const CRS& c, float* OUT, bool first) {
    int Mq = 8 * Tq, Mkv = 8 * Tkv;
    lnorm(Sq, c.ln1, XN, Mq);
    lnorm(Skv, c.ln2, XN2, Mkv);
    gemm(XN,  c.a.q, 768, 768, Mq,  MODE_BF, nullptr, nullptr, Qb);
    gemm(XN2, c.a.k, 768, 768, Mkv, MODE_BF, nullptr, nullptr, Kb);
    gemm(XN2, c.a.v, 768, 768, Mkv, MODE_BF, nullptr, nullptr, Vb);
    attn(Qb, Kb, Vb, XN, Tq, Tkv);
    // first: OUT = Sq + 0.5*(mha); second: OUT = OUT + 0.5*(mha)
    gemm(XN, c.a.o, 768, 768, Mq, MODE_HALFRES, first ? Sq : OUT, OUT, nullptr);
  };

  auto finffn = [&](float* OUT, int T, const FFNP& f, const LNP& n) {
    int M = 8 * T;
    lnorm(OUT, n, XN, M);
    gemm(XN, f.l1, 768, 3072, M, MODE_GELU, nullptr, nullptr, H1);
    gemm(H1, f.l2, 3072, 768, M, MODE_RES, OUT, OUT, nullptr);
  };

  // residual streams start as the inputs (inputs must not be mutated)
  (void)hipMemcpyAsync(Sv, visual, vN * 4, hipMemcpyDeviceToDevice, stream);
  (void)hipMemcpyAsync(Sa, audio,  vN * 4, hipMemcpyDeviceToDevice, stream);
  (void)hipMemcpyAsync(Sh, hr,     hN * 4, hipMemcpyDeviceToDevice, stream);

  encoder(Sv, 512, sav);
  encoder(Sa, 512, saa);
  encoder(Sh, 256, sah);

  float* vO = (float*)d_out;
  float* aO = vO + vN;
  float* hO = aO + vN;

  cross(Sv, 512, Sa, 512, v2a, vO, true);
  cross(Sv, 512, Sh, 256, v2h, vO, false);
  cross(Sa, 512, Sv, 512, a2v, aO, true);
  cross(Sa, 512, Sh, 256, a2h, aO, false);
  cross(Sh, 256, Sv, 512, h2v, hO, true);
  cross(Sh, 256, Sa, 512, h2a, hO, false);

  finffn(vO, 512, fv, nv);
  finffn(aO, 512, fa, na);
  finffn(hO, 256, fh, nh);
}
